// CSA_53566832115807
// MI455X (gfx1250) — matrix operand order repaired and checked by reading, not compiled, not run on hardware
//
#include <hip/hip_runtime.h>
#include <hip/hip_bf16.h>
#include <stdint.h>

// ---------------- problem constants ----------------
#define BS_   8
#define C_    256
#define H_    64
#define W_    64
#define E_    16
#define HID_  1024
#define S_    (BS_*H_*W_)          // 32768 tokens
#define CAP_  4096                 // int(2*S/E)
#define THRESH_     0.2f
#define LOSS_COEF_  0.01f

typedef __attribute__((ext_vector_type(16))) __bf16        v16bf;
typedef __attribute__((ext_vector_type(8)))  float         v8f;
typedef __attribute__((ext_vector_type(8)))  unsigned int  v8u;
typedef int v4i_gcc __attribute__((vector_size(16)));      // matches builtin param type

__device__ __forceinline__ unsigned short f2bf(float f) {
  unsigned int u = __float_as_uint(f);
  u += 0x7FFFu + ((u >> 16) & 1u);          // round-to-nearest-even
  return (unsigned short)(u >> 16);
}
__device__ __forceinline__ float bf2f(unsigned short h) {
  return __uint_as_float(((unsigned int)h) << 16);
}

// ---------------- async global->LDS copy (16B per lane) ----------------
__device__ __forceinline__ void cp_async16(void* lds_dst, const void* gsrc) {
#if defined(__gfx1250__) && __has_builtin(__builtin_amdgcn_global_load_async_to_lds_b128)
  __builtin_amdgcn_global_load_async_to_lds_b128(
      (__attribute__((address_space(1))) v4i_gcc*)gsrc,
      (__attribute__((address_space(3))) v4i_gcc*)lds_dst,
      0, 0);
#else
  *(uint4*)lds_dst = *(const uint4*)gsrc;     // fallback: vmem load + ds store
#endif
}
__device__ __forceinline__ void wait_async0() {
#if defined(__gfx1250__) && __has_builtin(__builtin_amdgcn_s_wait_asynccnt)
  __builtin_amdgcn_s_wait_asynccnt(0);
#elif defined(__gfx1250__)
  asm volatile("s_wait_asynccnt 0" ::: "memory");
#endif
}

// ---------------- tokens: t[s][c] = x[b][c][hw] + audio[b][c], bf16 ----------------
__global__ void tok_kernel(const float* __restrict__ x, const float* __restrict__ audio,
                           unsigned short* __restrict__ tbf) {
  size_t i = (size_t)blockIdx.x * blockDim.x + threadIdx.x;   // over S*C
  int c  = (int)(i & (C_ - 1));
  size_t s = i >> 8;
  int b  = (int)(s / (H_ * W_));
  int hw = (int)(s % (H_ * W_));
  float v = x[((size_t)b * C_ + c) * (H_ * W_) + hw] + audio[(size_t)b * C_ + c];
  tbf[i] = f2bf(v);
}

// ---------------- weight transpose+convert: w [E][D1][D2] f32 -> wt [E][D2][D1] bf16 ----------------
__global__ void twz_kernel(const float* __restrict__ w, unsigned short* __restrict__ wt,
                           int D1, int D2) {
  size_t i = (size_t)blockIdx.x * blockDim.x + threadIdx.x;   // over E*D1*D2
  int d2 = (int)(i % D2);
  size_t r = i / D2;
  int d1 = (int)(r % D1);
  int e  = (int)(r / D1);
  wt[((size_t)e * D2 + d2) * D1 + d1] = f2bf(w[i]);
}

// ---------------- router: softmax(t@wg), top-2 + threshold, gates, prob sums ----------------
__global__ void router_kernel(const unsigned short* __restrict__ tbf,
                              const float* __restrict__ wg,   // [C][E]
                              int* __restrict__ idx1, int* __restrict__ idx2,
                              float* __restrict__ g1o, float* __restrict__ g2o,
                              float* __restrict__ k2o, float* __restrict__ probsum) {
  __shared__ float swg[C_ * E_];     // 16 KB
  __shared__ float sps[E_];
  for (int i = threadIdx.x; i < C_ * E_; i += blockDim.x) swg[i] = wg[i];
  if (threadIdx.x < E_) sps[threadIdx.x] = 0.f;
  __syncthreads();

  int s = blockIdx.x * blockDim.x + threadIdx.x;    // S divisible by blockDim
  const unsigned short* tr = tbf + (size_t)s * C_;
  float logit[E_];
#pragma unroll
  for (int e = 0; e < E_; ++e) logit[e] = 0.f;
  for (int c = 0; c < C_; ++c) {
    float tv = bf2f(tr[c]);
    const float* wr = &swg[c * E_];
#pragma unroll
    for (int e = 0; e < E_; ++e) logit[e] += tv * wr[e];
  }
  float m = logit[0];
#pragma unroll
  for (int e = 1; e < E_; ++e) m = fmaxf(m, logit[e]);
  float pr[E_]; float sum = 0.f;
#pragma unroll
  for (int e = 0; e < E_; ++e) { pr[e] = expf(logit[e] - m); sum += pr[e]; }
  float inv = 1.f / sum;
#pragma unroll
  for (int e = 0; e < E_; ++e) pr[e] *= inv;

  int i1 = 0; float p1 = pr[0];
#pragma unroll
  for (int e = 1; e < E_; ++e) if (pr[e] > p1) { p1 = pr[e]; i1 = e; }
  int i2 = (i1 == 0) ? 1 : 0; float p2 = pr[i2];
#pragma unroll
  for (int e = 0; e < E_; ++e) if (e != i1 && pr[e] > p2) { p2 = pr[e]; i2 = e; }

  float k2 = (p2 > THRESH_) ? 1.f : 0.f;
  float denom = p1 + p2 * k2 + 1e-9f;
  idx1[s] = i1; idx2[s] = i2;
  g1o[s] = p1 / denom; g2o[s] = p2 * k2 / denom; k2o[s] = k2;

#pragma unroll
  for (int e = 0; e < E_; ++e) atomicAdd(&sps[e], pr[e]);
  __syncthreads();
  if (threadIdx.x < E_) atomicAdd(&probsum[threadIdx.x], sps[threadIdx.x]);
}

// ---------------- ordered per-expert position assignment (capacity cumsum) ----------------
__global__ void pos_kernel(const int* __restrict__ idx1, const int* __restrict__ idx2,
                           const float* __restrict__ k2, int* __restrict__ pos1,
                           int* __restrict__ pos2, int* __restrict__ count1) {
  int e = threadIdx.x;     // blockDim = 16, one block
  int cnt = 0;
  for (int s = 0; s < S_; ++s)
    if (idx1[s] == e) pos1[s] = cnt++;
  count1[e] = cnt;
  __syncthreads();
  for (int s = 0; s < S_; ++s)
    if (idx2[s] == e && k2[s] > 0.5f) pos2[s] = cnt++;
}

// ---------------- dispatch scatter into expert buffers (unique slots) ----------------
__global__ void disp_kernel(const unsigned short* __restrict__ tbf,
                            const int* __restrict__ idx1, const int* __restrict__ idx2,
                            const int* __restrict__ pos1, const int* __restrict__ pos2,
                            const float* __restrict__ k2,
                            unsigned short* __restrict__ buf) {
  int s = blockIdx.x;            // S blocks of 256 threads
  int c = threadIdx.x;
  unsigned short v = tbf[(size_t)s * C_ + c];
  int p1 = pos1[s];
  if (p1 < CAP_) buf[((size_t)idx1[s] * CAP_ + p1) * C_ + c] = v;
  if (k2[s] > 0.5f) {
    int p2 = pos2[s];
    if (p2 < CAP_) buf[((size_t)idx2[s] * CAP_ + p2) * C_ + c] = v;
  }
}

// ---------------- LDS-tiled WMMA GEMM ----------------
// Per expert:  A [CAP_ x lda] (row-major, K = lda-dim slice), Bt [N x ldb] (B column-major),
// out [CAP_ x N].  Block tile 64(M) x 128(N), K-step 32, double-buffered LDS, async staging.
// 8 waves arranged 2(M) x 4(N); each wave computes 32x32 via 4 WMMA accumulators.
template <bool RELU_BF16>
__global__ __launch_bounds__(256) void gemm_kernel(
    const unsigned short* __restrict__ Abase, int lda,
    const unsigned short* __restrict__ Btbase, int ldb,
    const float* __restrict__ bias,           // [E][N]
    void* __restrict__ outbase, int N, int K) {
  constexpr int BM = 64, BN = 128, BK = 32;
  __shared__ __align__(16) unsigned short sA[2][BM * BK];   // 2 x 4 KB
  __shared__ __align__(16) unsigned short sB[2][BN * BK];   // 2 x 8 KB

  const int nBlkN = N / BN;
  const int blocksPerE = (CAP_ / BM) * nBlkN;
  const int e  = blockIdx.x / blocksPerE;
  const int r  = blockIdx.x % blocksPerE;
  const int bm = r / nBlkN, bn = r % nBlkN;
  const unsigned short* A  = Abase  + ((size_t)e * CAP_ + (size_t)bm * BM) * lda;
  const unsigned short* Bt = Btbase + ((size_t)e * N    + (size_t)bn * BN) * ldb;

  const int tid = threadIdx.x;
  const int wave = tid >> 5;
  const int wm = (wave >> 2) * 32;          // 0 / 32
  const int wn = (wave & 3) * 32;           // 0 / 32 / 64 / 96
  const int lane = tid & 31, half = lane >> 4, lr = lane & 15;

  auto stage = [&](int sb, int k0) {
    // A tile: 64 rows x 64B -> 256 x 16B chunks, one per thread
    {
      int mrow = tid >> 2, seg = tid & 3;
      cp_async16(&sA[sb][mrow * BK + seg * 8],
                 A + (size_t)mrow * lda + k0 + seg * 8);
    }
    // B tile: 128 rows x 64B -> 512 x 16B chunks, two per thread
#pragma unroll
    for (int i = 0; i < 2; ++i) {
      int t = tid + i * 256;
      int nrow = t >> 2, seg = t & 3;
      cp_async16(&sB[sb][nrow * BK + seg * 8],
                 Bt + (size_t)nrow * ldb + k0 + seg * 8);
    }
  };

  v8f acc[2][2] = {};
  stage(0, 0);
  wait_async0();
  __syncthreads();

  const int KS = K / BK;
  for (int ks = 0; ks < KS; ++ks) {
    const int cur = ks & 1;
    if (ks + 1 < KS) stage(cur ^ 1, (ks + 1) * BK);

    v8u afr[2], bfr[2];
#pragma unroll
    for (int i = 0; i < 2; ++i) {
      // A fragment: ISA 16-bit A layout (VGPR0-3: K 0..15, VGPR4-7: K 16..31, +8*half)
      const unsigned int* ar = (const unsigned int*)&sA[cur][(wm + i * 16 + lr) * BK];
      uint4 lo = *(const uint4*)(ar + 4 * half);
      uint4 hi = *(const uint4*)(ar + 8 + 4 * half);
      afr[i][0] = lo.x; afr[i][1] = lo.y; afr[i][2] = lo.z; afr[i][3] = lo.w;
      afr[i][4] = hi.x; afr[i][5] = hi.y; afr[i][6] = hi.z; afr[i][7] = hi.w;
      // B fragment: same layout as A (VGPR0-3: K 0..15, VGPR4-7: K 16..31, +8*half); N = lane%16
      const unsigned int* br = (const unsigned int*)&sB[cur][(wn + i * 16 + lr) * BK];
      uint4 b0 = *(const uint4*)(br + 4 * half);
      uint4 b1 = *(const uint4*)(br + 8 + 4 * half);
      bfr[i][0] = b0.x; bfr[i][1] = b0.y; bfr[i][2] = b0.z; bfr[i][3] = b0.w;
      bfr[i][4] = b1.x; bfr[i][5] = b1.y; bfr[i][6] = b1.z; bfr[i][7] = b1.w;
    }
#pragma unroll
    for (int i = 0; i < 2; ++i)
#pragma unroll
      for (int j = 0; j < 2; ++j)
        acc[i][j] = __builtin_amdgcn_wmma_f32_16x16x32_bf16(
            false, __builtin_bit_cast(v16bf, afr[i]),
            false, __builtin_bit_cast(v16bf, bfr[j]),
            (short)0, acc[i][j], false, false);

    wait_async0();          // next tile staged (no-op on last iter)
    __syncthreads();        // all waves done reading `cur`, next buffer visible
  }

  // epilogue
#pragma unroll
  for (int i = 0; i < 2; ++i) {
#pragma unroll
    for (int j = 0; j < 2; ++j) {
      int n = bn * BN + wn + j * 16 + lr;
      float bv = bias[(size_t)e * N + n];
#pragma unroll
      for (int rr = 0; rr < 8; ++rr) {
        int m = bm * BM + wm + i * 16 + rr + 8 * half;
        float v = acc[i][j][rr] + bv;
        size_t off = ((size_t)e * CAP_ + m) * N + n;
        if (RELU_BF16) {
          v = v > 0.f ? v : 0.01f * v;        // leaky_relu
          ((unsigned short*)outbase)[off] = f2bf(v);
        } else {
          ((float*)outbase)[off] = v;
        }
      }
    }
  }
}

// ---------------- combine (gather) ----------------
__global__ void comb_kernel(const float* __restrict__ eo,
                            const int* __restrict__ idx1, const int* __restrict__ idx2,
                            const int* __restrict__ pos1, const int* __restrict__ pos2,
                            const float* __restrict__ g1, const float* __restrict__ g2,
                            const float* __restrict__ k2, float* __restrict__ y) {
  int s = blockIdx.x, c = threadIdx.x;
  float acc = 0.f;
  int p1 = pos1[s];
  if (p1 < CAP_) acc += g1[s] * eo[((size_t)idx1[s] * CAP_ + p1) * C_ + c];
  if (k2[s] > 0.5f) {
    int p2 = pos2[s];
    if (p2 < CAP_) acc += g2[s] * eo[((size_t)idx2[s] * CAP_ + p2) * C_ + c];
  }
  y[(size_t)s * C_ + c] = acc;
}

// ---------------- final 2-way gate + NCHW output ----------------
__global__ void fin_kernel(const float* __restrict__ yc, const float* __restrict__ ys,
                           const float* __restrict__ gcw, const float* __restrict__ gcb,
                           const float* __restrict__ gsw, const float* __restrict__ gsb,
                           float* __restrict__ out) {
  int s = blockIdx.x * blockDim.x + threadIdx.x;   // token
  int b  = s / (H_ * W_);
  int hw = s % (H_ * W_);
  const float* ycr = yc + (size_t)s * C_;
  const float* ysr = ys + (size_t)s * C_;
  float ac = gcb[0], as = gsb[0];
  for (int c = 0; c < C_; ++c) {
    ac += ycr[c] * gcw[c] + ysr[c] * gcw[C_ + c];
    as += ycr[c] * gsw[c] + ysr[c] * gsw[C_ + c];
  }
  float m = fmaxf(ac, as);
  float e0 = expf(ac - m), e1 = expf(as - m);
  float inv = 1.f / (e0 + e1);
  float wA = e0 * inv, wB = e1 * inv;
  for (int c = 0; c < C_; ++c)
    out[((size_t)b * C_ + c) * (H_ * W_) + hw] = ycr[c] * wA + ysr[c] * wB;
}

// ---------------- aux loss ----------------
__global__ void loss_kernel(const int* __restrict__ cnt_c, const float* __restrict__ ps_c,
                            const int* __restrict__ cnt_s, const float* __restrict__ ps_s,
                            float* __restrict__ out_loss) {
  if (threadIdx.x == 0 && blockIdx.x == 0) {
    float lc = 0.f, ls = 0.f;
    for (int e = 0; e < E_; ++e) {
      lc += ((float)cnt_c[e] / (float)S_) * (ps_c[e] / (float)S_);
      ls += ((float)cnt_s[e] / (float)S_) * (ps_s[e] / (float)S_);
    }
    lc *= (float)E_ * LOSS_COEF_;
    ls *= (float)E_ * LOSS_COEF_;
    out_loss[0] = 0.5f * (lc + ls);
  }
}

// ---------------- host orchestration ----------------
static void run_moe(const float* wg, const float* w1, const float* b1,
                    const float* w2, const float* b2,
                    unsigned short* tbf, unsigned short* w1t, unsigned short* w2t,
                    unsigned short* buf, unsigned short* h, float* eo,
                    int* idx1, int* idx2, int* pos1, int* pos2,
                    float* g1, float* g2, float* k2,
                    int* cnt, float* ps, float* y, hipStream_t stream) {
  (void)hipMemsetAsync(ps, 0, E_ * sizeof(float), stream);
  router_kernel<<<S_ / 256, 256, 0, stream>>>(tbf, wg, idx1, idx2, g1, g2, k2, ps);
  pos_kernel<<<1, E_, 0, stream>>>(idx1, idx2, k2, pos1, pos2, cnt);
  (void)hipMemsetAsync(buf, 0, (size_t)E_ * CAP_ * C_ * sizeof(unsigned short), stream);
  disp_kernel<<<S_, C_, 0, stream>>>(tbf, idx1, idx2, pos1, pos2, k2, buf);
  twz_kernel<<<(E_ * C_ * HID_) / 256, 256, 0, stream>>>(w1, w1t, C_, HID_);
  twz_kernel<<<(E_ * C_ * HID_) / 256, 256, 0, stream>>>(w2, w2t, HID_, C_);
  gemm_kernel<true><<<E_ * (CAP_ / 64) * (HID_ / 128), 256, 0, stream>>>(
      buf, C_, w1t, C_, b1, (void*)h, HID_, C_);
  gemm_kernel<false><<<E_ * (CAP_ / 64) * (C_ / 128), 256, 0, stream>>>(
      h, HID_, w2t, HID_, b2, (void*)eo, C_, HID_);
  comb_kernel<<<S_, C_, 0, stream>>>(eo, idx1, idx2, pos1, pos2, g1, g2, k2, y);
}

extern "C" void kernel_launch(void* const* d_in, const int* in_sizes, int n_in,
                              void* d_out, int out_size, void* d_ws, size_t ws_size,
                              hipStream_t stream) {
  const float* x      = (const float*)d_in[0];
  const float* audio  = (const float*)d_in[1];   // [BS,1,C] -> mean over axis 1 == identity
  const float* wg_c   = (const float*)d_in[2];
  const float* w1_c   = (const float*)d_in[3];
  const float* b1_c   = (const float*)d_in[4];
  const float* w2_c   = (const float*)d_in[5];
  const float* b2_c   = (const float*)d_in[6];
  const float* wg_s   = (const float*)d_in[7];
  const float* w1_s   = (const float*)d_in[8];
  const float* b1_s   = (const float*)d_in[9];
  const float* w2_s   = (const float*)d_in[10];
  const float* b2_s   = (const float*)d_in[11];
  const float* gcw    = (const float*)d_in[12];
  const float* gcb    = (const float*)d_in[13];
  const float* gsw    = (const float*)d_in[14];
  const float* gsb    = (const float*)d_in[15];
  float* out = (float*)d_out;

  // workspace carve-up
  char* p = (char*)d_ws;
  size_t off = 0;
  auto take = [&](size_t bytes) -> char* {
    char* r = p + off;
    off = (off + bytes + 255) & ~(size_t)255;
    return r;
  };
  unsigned short* tbf = (unsigned short*)take((size_t)S_ * C_ * 2);
  unsigned short* w1t = (unsigned short*)take((size_t)E_ * HID_ * C_ * 2);
  unsigned short* w2t = (unsigned short*)take((size_t)E_ * C_ * HID_ * 2);
  unsigned short* buf = (unsigned short*)take((size_t)E_ * CAP_ * C_ * 2);
  unsigned short* hbf = (unsigned short*)take((size_t)E_ * CAP_ * HID_ * 2);
  float* eo  = (float*)take((size_t)E_ * CAP_ * C_ * 4);
  float* y_c = (float*)take((size_t)S_ * C_ * 4);
  float* y_s = (float*)take((size_t)S_ * C_ * 4);
  int*   idx1 = (int*)take((size_t)S_ * 4);
  int*   idx2 = (int*)take((size_t)S_ * 4);
  int*   pos1 = (int*)take((size_t)S_ * 4);
  int*   pos2 = (int*)take((size_t)S_ * 4);
  float* g1   = (float*)take((size_t)S_ * 4);
  float* g2   = (float*)take((size_t)S_ * 4);
  float* k2   = (float*)take((size_t)S_ * 4);
  int*   cnt_c = (int*)take(E_ * 4);
  int*   cnt_s = (int*)take(E_ * 4);
  float* ps_c  = (float*)take(E_ * 4);
  float* ps_s  = (float*)take(E_ * 4);

  tok_kernel<<<(S_ * C_) / 256, 256, 0, stream>>>(x, audio, tbf);

  run_moe(wg_c, w1_c, b1_c, w2_c, b2_c, tbf, w1t, w2t, buf, hbf, eo,
          idx1, idx2, pos1, pos2, g1, g2, k2, cnt_c, ps_c, y_c, stream);
  run_moe(wg_s, w1_s, b1_s, w2_s, b2_s, tbf, w1t, w2t, buf, hbf, eo,
          idx1, idx2, pos1, pos2, g1, g2, k2, cnt_s, ps_s, y_s, stream);

  fin_kernel<<<S_ / 256, 256, 0, stream>>>(y_c, y_s, gcw, gcb, gsw, gsb, out);
  loss_kernel<<<1, 1, 0, stream>>>(cnt_c, ps_c, cnt_s, ps_s, out + (size_t)BS_ * C_ * H_ * W_);
}
